// InstrumentedEchoChamber_75892072121006
// MI455X (gfx1250) — matrix operand order repaired and checked by reading, not compiled, not run on hardware
//
#include <hip/hip_runtime.h>

// ---------------- problem constants ----------------
#define TT 512
#define BB 64
#define DD 1024
#define HH 8
#define DH 128
#define MM (TT * BB)            // 32768 rows
#define LN_EPS 1e-5f

// ---------------- vector typedefs ----------------
typedef __attribute__((ext_vector_type(16))) __bf16 v16bf;
typedef __attribute__((ext_vector_type(8)))  float  v8f;
typedef __attribute__((ext_vector_type(4)))  __bf16 bf16x4;
typedef __attribute__((ext_vector_type(4)))  unsigned int u32x4;
typedef __attribute__((ext_vector_type(8)))  unsigned int u32x8;
typedef __attribute__((ext_vector_type(4)))  int i32x4;

// address-space-qualified types (force global_* / ds_* selection)
typedef __attribute__((address_space(1))) float gfloat;
typedef __attribute__((address_space(1))) const u32x4 gc_u32x4;
typedef __attribute__((address_space(1))) i32x4 g_i32x4;
typedef __attribute__((address_space(3))) i32x4 l_i32x4;

// ---- 16B global -> LDS copy: async DMA if available, else global+ds ----
__device__ __forceinline__ void async_copy16(const __bf16* g, __bf16* l) {
#if __has_builtin(__builtin_amdgcn_global_load_async_to_lds_b128)
  __builtin_amdgcn_global_load_async_to_lds_b128(
      (g_i32x4*)(i32x4*)(__bf16*)g, (l_i32x4*)(i32x4*)l, 0, 0);
#else
  *(u32x4*)l = *(const gc_u32x4*)g;
#endif
}

__device__ __forceinline__ void wait_async0() {
#if __has_builtin(__builtin_amdgcn_s_wait_asynccnt)
  __builtin_amdgcn_s_wait_asynccnt(0);
#else
  asm volatile("s_wait_asynccnt 0x0" ::: "memory");
#endif
}

// =======================================================================
// f32 -> bf16 conversion (4 elems / thread)
// =======================================================================
__global__ __launch_bounds__(256) void cvt_bf16_kernel(
    const float* __restrict__ src, __bf16* __restrict__ dst, int n4) {
  int i = blockIdx.x * 256 + threadIdx.x;
  if (i >= n4) return;
  float4 v = ((const float4*)src)[i];
  bf16x4 o;
  o[0] = (__bf16)v.x; o[1] = (__bf16)v.y; o[2] = (__bf16)v.z; o[3] = (__bf16)v.w;
  ((bf16x4*)dst)[i] = o;
}

// Wc1 = Wr + Wi, Wc2 = Wr - Wi   (bf16 outputs)
__global__ __launch_bounds__(256) void prep_wc_kernel(
    const float* __restrict__ Wr, const float* __restrict__ Wi,
    __bf16* __restrict__ Wc1, __bf16* __restrict__ Wc2, int n4) {
  int i = blockIdx.x * 256 + threadIdx.x;
  if (i >= n4) return;
  float4 r = ((const float4*)Wr)[i];
  float4 m = ((const float4*)Wi)[i];
  bf16x4 a, b;
  a[0] = (__bf16)(r.x + m.x); a[1] = (__bf16)(r.y + m.y);
  a[2] = (__bf16)(r.z + m.z); a[3] = (__bf16)(r.w + m.w);
  b[0] = (__bf16)(r.x - m.x); b[1] = (__bf16)(r.y - m.y);
  b[2] = (__bf16)(r.z - m.z); b[3] = (__bf16)(r.w - m.w);
  ((bf16x4*)Wc1)[i] = a;
  ((bf16x4*)Wc2)[i] = b;
}

// bc = br + bi (f32, length D)
__global__ __launch_bounds__(256) void bias_sum_kernel(
    const float* __restrict__ br, const float* __restrict__ bi,
    float* __restrict__ bc) {
  int i = blockIdx.x * 256 + threadIdx.x;
  if (i < DD) bc[i] = br[i] + bi[i];
}

// =======================================================================
// Generic bf16 WMMA GEMM:  C[M,N] = sum_p A_p[M,K] * W_p[N,K]^T  (+ bias[N])
// Block tile 128x128, BK=32. 8 waves in a 4x2 grid, wave tile 32x64
// (2x4 of 16x16 WMMA -> 8 v_wmma per K-step).
// Double-buffered LDS fed by async global->LDS DMA; one barrier +
// one s_wait_asynccnt per K-step; copy of tile k+1 overlaps WMMA of tile k.
// =======================================================================
__global__ __launch_bounds__(256) void wmma_gemm_kernel(
    const __bf16* __restrict__ A0, const __bf16* __restrict__ W0,
    const __bf16* __restrict__ A1, const __bf16* __restrict__ W1,
    const float* __restrict__ bias, float* __restrict__ C,
    int N, int K) {
  constexpr int BM = 128, BN = 128, BK = 32;
  constexpr int LDA = 40;  // bf16 elems per row (80B, 16B-aligned, padded)
  __shared__ __align__(16) __bf16 As[2][BM * LDA];
  __shared__ __align__(16) __bf16 Bs[2][BN * LDA];

  const int tid   = threadIdx.x;
  const int lane  = tid & 31;
  const int wv    = tid >> 5;        // 0..7
  const int mw    = wv >> 1;         // 0..3  -> M offset mw*32
  const int nw    = wv & 1;          // 0..1  -> N offset nw*64
  const int laneM = lane & 15;
  const int half  = lane >> 4;

  const int row0 = blockIdx.y * BM;
  const int n0   = blockIdx.x * BN;

  const int npass = (A1 != nullptr) ? 2 : 1;
  const int nkt   = K / BK;
  const int total = npass * nkt;

  // issue all async copies for one 128x32 A tile + 128x32 B tile
  auto issue = [&](int tile, int buf) {
    const __bf16* Ap = (tile < nkt) ? A0 : A1;
    const __bf16* Wp = (tile < nkt) ? W0 : W1;
    const int k0 = ((tile < nkt) ? tile : tile - nkt) * BK;
    __bf16* Ad = &As[buf][0];
    __bf16* Bd = &Bs[buf][0];
    #pragma unroll
    for (int i = 0; i < 2; ++i) {
      int c  = tid + i * 256;          // 0..511
      int m  = c >> 2;                 // 0..127
      int kc = (c & 3) * 8;            // 0,8,16,24
      async_copy16(Ap + (size_t)(row0 + m) * K + k0 + kc, Ad + m * LDA + kc);
      async_copy16(Wp + (size_t)(n0 + m) * K + k0 + kc,  Bd + m * LDA + kc);
    }
  };

  v8f acc[2][4] = {};

  issue(0, 0);
  for (int t = 0; t < total; ++t) {
    wait_async0();       // this wave's copies for tile t have landed in LDS
    __syncthreads();     // everyone's copies landed; prior readers done
    if (t + 1 < total) issue(t + 1, (t + 1) & 1);  // overlap with WMMAs below

    const __bf16* Ab = &As[t & 1][0];
    const __bf16* Bb = &Bs[t & 1][0];

    // ---- LDS -> fragments (ISA 16-bit A 16x32 / B 32x16 layouts) ----
    v16bf af[2], bfr[4];
    #pragma unroll
    for (int mi = 0; mi < 2; ++mi) {
      int mrow = mw * 32 + mi * 16 + laneM;
      u32x8 v;
      v.lo = *(const u32x4*)(Ab + mrow * LDA + half * 8);       // K = half*8 ..+7
      v.hi = *(const u32x4*)(Ab + mrow * LDA + 16 + half * 8);  // K = 16+half*8 ..+7
      af[mi] = __builtin_bit_cast(v16bf, v);
    }
    #pragma unroll
    for (int ni = 0; ni < 4; ++ni) {
      int nrow = nw * 64 + ni * 16 + laneM;
      u32x8 v;
      v.lo = *(const u32x4*)(Bb + nrow * LDA + half * 8);       // K = half*8 ..+7
      v.hi = *(const u32x4*)(Bb + nrow * LDA + 16 + half * 8);  // K = 16+half*8 ..+7
      bfr[ni] = __builtin_bit_cast(v16bf, v);
    }
    // ---- 8 WMMAs ----
    #pragma unroll
    for (int mi = 0; mi < 2; ++mi)
      #pragma unroll
      for (int ni = 0; ni < 4; ++ni)
        acc[mi][ni] = __builtin_amdgcn_wmma_f32_16x16x32_bf16(
            false, af[mi], false, bfr[ni], (short)0, acc[mi][ni], false, false);
  }

  // ---- epilogue: C/D f32 layout -> global f32 (+bias) ----
  gfloat* Cg = (gfloat*)C;
  const gfloat* biasg = (const gfloat*)bias;
  #pragma unroll
  for (int mi = 0; mi < 2; ++mi) {
    #pragma unroll
    for (int ni = 0; ni < 4; ++ni) {
      int col   = n0 + nw * 64 + ni * 16 + laneM;
      float b   = bias ? biasg[col] : 0.0f;
      int rbase = row0 + mw * 32 + mi * 16 + half * 8;
      #pragma unroll
      for (int j = 0; j < 8; ++j)
        Cg[(size_t)(rbase + j) * N + col] = acc[mi][ni][j] + b;
    }
  }
}

// =======================================================================
// LayerNorm over D=1024, one block (256 thr) per row; writes bf16
// =======================================================================
__global__ __launch_bounds__(256) void layernorm_kernel(
    const float* __restrict__ xc, const float* __restrict__ gamma,
    const float* __restrict__ beta, __bf16* __restrict__ out) {
  __shared__ float rs[256], rs2[256];
  const int row = blockIdx.x;
  const int tid = threadIdx.x;
  float4 v = ((const float4*)(xc + (size_t)row * DD))[tid];
  float s  = v.x + v.y + v.z + v.w;
  float s2 = v.x * v.x + v.y * v.y + v.z * v.z + v.w * v.w;
  rs[tid] = s; rs2[tid] = s2;
  __syncthreads();
  for (int off = 128; off > 0; off >>= 1) {
    if (tid < off) { rs[tid] += rs[tid + off]; rs2[tid] += rs2[tid + off]; }
    __syncthreads();
  }
  float mu  = rs[0] * (1.0f / DD);
  float var = rs2[0] * (1.0f / DD) - mu * mu;
  float inv = rsqrtf(var + LN_EPS);
  float4 g = ((const float4*)gamma)[tid];
  float4 b = ((const float4*)beta)[tid];
  bf16x4 o;
  o[0] = (__bf16)((v.x - mu) * inv * g.x + b.x);
  o[1] = (__bf16)((v.y - mu) * inv * g.y + b.y);
  o[2] = (__bf16)((v.z - mu) * inv * g.z + b.z);
  o[3] = (__bf16)((v.w - mu) * inv * g.w + b.w);
  ((bf16x4*)(out + (size_t)row * DD))[tid] = o;
}

// =======================================================================
// Gate: one wave32 per (row, head): sigmoid( q . k / sqrt(DH) )
// =======================================================================
__global__ __launch_bounds__(256) void gate_kernel(
    const float* __restrict__ qkv, float* __restrict__ gate) {
  const int wid  = blockIdx.x * 8 + (threadIdx.x >> 5);  // = row*8 + h
  const int lane = threadIdx.x & 31;
  const int row  = wid >> 3;
  const int h    = wid & 7;
  const float* qp = qkv + (size_t)row * (3 * DD) + h * DH + lane * 4;
  float4 q = *(const float4*)qp;
  float4 k = *(const float4*)(qp + DD);
  float p = q.x * k.x + q.y * k.y + q.z * k.z + q.w * k.w;
  #pragma unroll
  for (int m = 16; m > 0; m >>= 1) p += __shfl_xor(p, m, 32);
  if (lane == 0) {
    float logit = p * 0.08838834764831845f;  // 1/sqrt(128)
    gate[wid] = 1.0f / (1.0f + __expf(-logit));
  }
}

// =======================================================================
// EMA scan over T (only sequential axis). One thread per (b,h,dh) chain.
// retrieved (bf16) = gate * state; also writes final_state.
// =======================================================================
__global__ __launch_bounds__(256) void scan_kernel(
    const float* __restrict__ qkv, const float* __restrict__ gate,
    const float* __restrict__ v0, const float* __restrict__ log_alpha,
    __bf16* __restrict__ retr, float* __restrict__ final_state) {
  const int idx = blockIdx.x * 256 + threadIdx.x;  // 0..65535 = b*1024 + h*128 + dh
  const int rem = idx & (DD - 1);
  const int b   = idx >> 10;
  const int h   = rem >> 7;
  const float alpha = 1.0f / (1.0f + __expf(-log_alpha[h]));
  const float beta  = 1.0f - alpha;
  float st = v0[idx];
  for (int t = 0; t < TT; ++t) {
    const size_t row = (size_t)t * BB + b;
    float vv = qkv[row * (3 * DD) + 2 * DD + rem];
    float g  = gate[row * HH + h];
    st = alpha * vv + beta * st;
    retr[row * DD + rem] = (__bf16)(g * st);
  }
  final_state[idx] = st;
}

// =======================================================================
// Host orchestration
// =======================================================================
extern "C" void kernel_launch(void* const* d_in, const int* in_sizes, int n_in,
                              void* d_out, int out_size, void* d_ws, size_t ws_size,
                              hipStream_t stream) {
  (void)in_sizes; (void)n_in; (void)out_size; (void)ws_size;
  const float* x_real    = (const float*)d_in[0];
  const float* x_imag    = (const float*)d_in[1];
  const float* vstate0   = (const float*)d_in[2];
  // d_in[3] = t (unused)
  const float* Wr_in     = (const float*)d_in[4];
  const float* Wi_in     = (const float*)d_in[5];
  const float* br_in     = (const float*)d_in[6];
  const float* bi_in     = (const float*)d_in[7];
  const float* ln_gamma  = (const float*)d_in[8];
  const float* ln_beta   = (const float*)d_in[9];
  const float* W_qkv     = (const float*)d_in[10];
  const float* log_alpha = (const float*)d_in[11];
  const float* Wr_out    = (const float*)d_in[12];
  const float* Wi_out    = (const float*)d_in[13];

  // ---- workspace carve-out (256B aligned) ----
  char* w = (char*)d_ws;
  auto carve = [&](size_t bytes) -> char* {
    char* p = w;
    w += (bytes + 255) & ~(size_t)255;
    return p;
  };
  const size_t MK  = (size_t)MM * DD;        // 33,554,432
  __bf16* Xr_bf   = (__bf16*)carve(MK * 2);
  __bf16* Xi_bf   = (__bf16*)carve(MK * 2);
  __bf16* Wc1_bf  = (__bf16*)carve((size_t)DD * DD * 2);
  __bf16* Wc2_bf  = (__bf16*)carve((size_t)DD * DD * 2);
  __bf16* Wqkv_bf = (__bf16*)carve((size_t)3 * DD * DD * 2);
  __bf16* Wor_bf  = (__bf16*)carve((size_t)DD * DD * 2);
  __bf16* Woi_bf  = (__bf16*)carve((size_t)DD * DD * 2);
  float*  bc      = (float*) carve(DD * 4);
  float*  xc      = (float*) carve(MK * 4);
  __bf16* xcn_bf  = (__bf16*)carve(MK * 2);
  float*  qkv     = (float*) carve((size_t)MM * 3 * DD * 4);
  float*  gatebuf = (float*) carve((size_t)MM * HH * 4);
  __bf16* retr_bf = (__bf16*)carve(MK * 2);

  float* out_real    = (float*)d_out;
  float* out_imag    = out_real + MK;
  float* final_state = out_imag + MK;

  // ---- 1. pack inputs / weights to bf16 ----
  cvt_bf16_kernel<<<dim3((unsigned)(MK / 4 / 256)), 256, 0, stream>>>(x_real, Xr_bf, (int)(MK / 4));
  cvt_bf16_kernel<<<dim3((unsigned)(MK / 4 / 256)), 256, 0, stream>>>(x_imag, Xi_bf, (int)(MK / 4));
  prep_wc_kernel<<<dim3(DD * DD / 4 / 256), 256, 0, stream>>>(Wr_in, Wi_in, Wc1_bf, Wc2_bf, DD * DD / 4);
  cvt_bf16_kernel<<<dim3(3 * DD * DD / 4 / 256), 256, 0, stream>>>(W_qkv, Wqkv_bf, 3 * DD * DD / 4);
  cvt_bf16_kernel<<<dim3(DD * DD / 4 / 256), 256, 0, stream>>>(Wr_out, Wor_bf, DD * DD / 4);
  cvt_bf16_kernel<<<dim3(DD * DD / 4 / 256), 256, 0, stream>>>(Wi_out, Woi_bf, DD * DD / 4);
  bias_sum_kernel<<<dim3(DD / 256), 256, 0, stream>>>(br_in, bi_in, bc);

  // ---- 2. xc = Xr*Wc1^T + Xi*Wc2^T + bc ----
  wmma_gemm_kernel<<<dim3(DD / 128, MM / 128), 256, 0, stream>>>(
      Xr_bf, Wc1_bf, Xi_bf, Wc2_bf, bc, xc, DD, DD);

  // ---- 3. LayerNorm -> bf16 ----
  layernorm_kernel<<<dim3(MM), 256, 0, stream>>>(xc, ln_gamma, ln_beta, xcn_bf);

  // ---- 4. qkv = xcn * Wqkv^T ----
  wmma_gemm_kernel<<<dim3(3 * DD / 128, MM / 128), 256, 0, stream>>>(
      xcn_bf, Wqkv_bf, nullptr, nullptr, nullptr, qkv, 3 * DD, DD);

  // ---- 5. gate = sigmoid(q.k / sqrt(DH)) ----
  gate_kernel<<<dim3(MM), 256, 0, stream>>>(qkv, gatebuf);

  // ---- 6. EMA scan over T, fused gate multiply -> retrieved (bf16) ----
  scan_kernel<<<dim3(BB * DD / 256), 256, 0, stream>>>(
      qkv, gatebuf, vstate0, log_alpha, retr_bf, final_state);

  // ---- 7. output projections ----
  wmma_gemm_kernel<<<dim3(DD / 128, MM / 128), 256, 0, stream>>>(
      retr_bf, Wor_bf, nullptr, nullptr, nullptr, out_real, DD, DD);
  wmma_gemm_kernel<<<dim3(DD / 128, MM / 128), 256, 0, stream>>>(
      retr_bf, Woi_bf, nullptr, nullptr, nullptr, out_imag, DD, DD);
}
